// LSTM_NOG_24799141167553
// MI455X (gfx1250) — matrix operand order repaired and checked by reading, not compiled, not run on hardware
//
#include <hip/hip_runtime.h>
#include <hip/hip_bf16.h>
#include <math.h>

typedef __attribute__((ext_vector_type(16))) __bf16 v16bf;
typedef __attribute__((ext_vector_type(8)))  float  v8f;

#define BATCH     32768
#define D_IN      256
#define UNITS     512
#define KDIM      768        // D_IN + UNITS
#define NGATECOLS 1536       // 3 * UNITS (i, f, c gates)
#define W4U       2048       // 4 * UNITS (source weight row stride)
#define LDS_STRIDE 776       // 768 + 8 bf16 pad (keeps 16B alignment, spreads banks)

// Branch-free tanh: 1 - 2/(exp(2x)+1), with exp(2x) = exp2(x * 2*log2(e)).
// Lowers to v_mul + v_exp_f32 + v_add + v_rcp_f32 + v_fma; saturates
// correctly at +/-1 on exp overflow/underflow, no EXEC divergence.
__device__ __forceinline__ float fast_tanh(float x) {
    float t = __builtin_amdgcn_exp2f(x * 2.885390081777927f);  // 2*log2(e)
    return fmaf(-2.0f, __builtin_amdgcn_rcpf(t + 1.0f), 1.0f);
}

// ---------------------------------------------------------------------------
// Kernel 1: pack W_t[col][k] (bf16, column-major over the combined K=768 dim)
//   col in [0,1536): gate-major column g*512+u
//   k   in [0,768):  k<256 -> kernel[k][col], else recurrent_kernel[k-256][col]
// ---------------------------------------------------------------------------
__global__ void pack_weights(const float* __restrict__ kern,
                             const float* __restrict__ rkern,
                             __bf16* __restrict__ wt) {
    int idx = blockIdx.x * blockDim.x + threadIdx.x;   // = k*1536 + col
    if (idx >= KDIM * NGATECOLS) return;
    int k   = idx / NGATECOLS;
    int col = idx - k * NGATECOLS;
    float v = (k < D_IN) ? kern[(size_t)k * W4U + col]
                         : rkern[(size_t)(k - D_IN) * W4U + col];
    wt[(size_t)col * KDIM + k] = (__bf16)v;            // coalesced read, strided write
}

// ---------------------------------------------------------------------------
// Kernel 2: fused LSTM step.  Block tile: 64 batch rows x 64 units (3 gates).
// 8 waves as 2(M) x 4(N); each wave: 32 rows x 16 units x {i,f,c}.
// ---------------------------------------------------------------------------
__launch_bounds__(256)
__global__ void lstm_fused(const float* __restrict__ x,
                           const float* __restrict__ h,
                           const float* __restrict__ c,
                           const __bf16* __restrict__ wt,
                           const float* __restrict__ bias,
                           float* __restrict__ out) {
    __shared__ __bf16 As[64 * LDS_STRIDE];             // ~97 KB

    const int m0  = blockIdx.x * 64;
    const int u0  = blockIdx.y * 64;
    const int tid = threadIdx.x;

    // ---- Stage A = [x | h] tile (64 x 768 fp32 -> bf16) into LDS ----------
    // float4 global loads, packed 4x bf16 (8B) LDS stores.
    {
        const float4* __restrict__ x4 = (const float4*)x;  // row stride 64 float4
        const float4* __restrict__ h4 = (const float4*)h;  // row stride 128 float4
        for (int i = tid; i < 64 * (KDIM / 4); i += 256) {
            int r  = i / (KDIM / 4);                   // KDIM/4 = 192
            int c4 = i - r * (KDIM / 4);               // [0,192)
            float4 v = (c4 < D_IN / 4)
                         ? x4[(size_t)(m0 + r) * (D_IN / 4) + c4]
                         : h4[(size_t)(m0 + r) * (UNITS / 4) + (c4 - D_IN / 4)];
            union { __bf16 b[4]; uint2 u; } p;
            p.b[0] = (__bf16)v.x; p.b[1] = (__bf16)v.y;
            p.b[2] = (__bf16)v.z; p.b[3] = (__bf16)v.w;
            *(uint2*)&As[r * LDS_STRIDE + c4 * 4] = p.u;   // 8B-aligned ds_store_b64
        }
    }
    __syncthreads();

    const int wave  = tid >> 5;
    const int lane  = tid & 31;
    const int waveM = wave >> 2;                       // 0..1 -> 32-row half
    const int waveN = wave & 3;                        // 0..3 -> 16-unit slice
    const int lhalf = lane >> 4;                       // 0 or 1
    const int l16   = lane & 15;

    const int unit  = u0 + waveN * 16 + l16;           // this lane's unit column

    // Accumulators: [row-frag 0/1][gate i/f/c]
    v8f acc[2][3];
#pragma unroll
    for (int rf = 0; rf < 2; ++rf)
#pragma unroll
        for (int g = 0; g < 3; ++g)
            acc[rf][g] = (v8f){0.f,0.f,0.f,0.f,0.f,0.f,0.f,0.f};

    // A-fragment LDS element bases (ISA 16-bit A layout):
    //   lane<16: K chunks {0..7, 16..23}; lane>=16: {8..15, 24..31}
    const int arow[2] = { (waveM * 32 +  0 + l16) * LDS_STRIDE,
                          (waveM * 32 + 16 + l16) * LDS_STRIDE };
    const int aoff = lhalf * 8;                        // element offset of first chunk

    // B-fragment global bases (uint4 index): lane<16 reads K {0..7,16..23}, lane>=16 K {8..15,24..31}
    const uint4* __restrict__ wtq = (const uint4*)wt;
    size_t bbase[3];
#pragma unroll
    for (int g = 0; g < 3; ++g)
        bbase[g] = (size_t)(g * UNITS + unit) * (KDIM / 8) + (size_t)(lhalf * 1);

    const uint4* __restrict__ lq = (const uint4*)As;

#pragma unroll 2
    for (int kb = 0; kb < KDIM; kb += 32) {
        // Load two A fragments from LDS (2 x ds_load_b128 each)
        v16bf a[2];
#pragma unroll
        for (int rf = 0; rf < 2; ++rf) {
            union { uint4 q[2]; v16bf v; } ua;
            int e0 = (arow[rf] + kb + aoff) >> 3;      // uint4 index
            ua.q[0] = lq[e0];
            ua.q[1] = lq[e0 + 2];                      // +16 elements
            a[rf] = ua.v;
        }
        // For each gate: load B fragment (2 x global_load_b128), 2 WMMAs
#pragma unroll
        for (int g = 0; g < 3; ++g) {
            union { uint4 q[2]; v16bf v; } ub;
            size_t bi = bbase[g] + (size_t)(kb >> 3);
            ub.q[0] = wtq[bi];
            ub.q[1] = wtq[bi + 2];                     // +16 elements
            v16bf b = ub.v;
            acc[0][g] = __builtin_amdgcn_wmma_f32_16x16x32_bf16(
                false, a[0], false, b, (short)0, acc[0][g], false, false);
            acc[1][g] = __builtin_amdgcn_wmma_f32_16x16x32_bf16(
                false, a[1], false, b, (short)0, acc[1][g], false, false);
        }
    }

    // ---- Epilogue: gates -> c_new, h_new --------------------------------
    const float bi_ = bias[unit];
    const float bf_ = bias[UNITS + unit];
    const float bc_ = bias[2 * UNITS + unit];
    const size_t outHalf = (size_t)BATCH * UNITS;

#pragma unroll
    for (int rf = 0; rf < 2; ++rf) {
        const int mBase = m0 + waveM * 32 + rf * 16 + lhalf * 8;
#pragma unroll
        for (int v = 0; v < 8; ++v) {
            const int m = mBase + v;                   // C layout: row = v + 8*lhalf
            float zi = acc[rf][0][v] + bi_;
            float zf = acc[rf][1][v] + bf_;
            float zc = acc[rf][2][v] + bc_;
            float ig = fminf(fmaxf(0.2f * zi + 0.5f, 0.0f), 1.0f);
            float fg = fminf(fmaxf(0.2f * zf + 0.5f, 0.0f), 1.0f);
            float ct = fast_tanh(zc);
            float cold = c[(size_t)m * UNITS + unit];
            float cn = fg * cold + ig * ct;
            float hn = fast_tanh(cn);
            out[(size_t)m * UNITS + unit] = hn;                // h_new
            out[outHalf + (size_t)m * UNITS + unit] = cn;      // c_new
        }
    }
}

// ---------------------------------------------------------------------------
extern "C" void kernel_launch(void* const* d_in, const int* in_sizes, int n_in,
                              void* d_out, int out_size, void* d_ws, size_t ws_size,
                              hipStream_t stream) {
    const float* x     = (const float*)d_in[0];   // inputs [B, 256]
    const float* h     = (const float*)d_in[1];   // h      [B, 512]
    const float* c     = (const float*)d_in[2];   // c      [B, 512]
    const float* kern  = (const float*)d_in[3];   // kernel [256, 2048]
    const float* rkern = (const float*)d_in[4];   // recurrent_kernel [512, 2048]
    const float* bias  = (const float*)d_in[5];   // bias [2048]
    float* out = (float*)d_out;                   // [h_new | c_new], each [B,512]
    __bf16* wt = (__bf16*)d_ws;                   // W_t [1536][768] bf16 (2.25 MB)

    pack_weights<<<(KDIM * NGATECOLS) / 256, 256, 0, stream>>>(kern, rkern, wt);

    dim3 grid(BATCH / 64, UNITS / 64);
    lstm_fused<<<grid, 256, 0, stream>>>(x, h, c, wt, bias, out);
}
